// Attention_71768903516546
// MI455X (gfx1250) — matrix operand order repaired and checked by reading, not compiled, not run on hardware
//
#include <hip/hip_runtime.h>
#include <math.h>

#define B_LEN 256
#define S_LEN 515
#define D_LEN 512
#define SOUT  512      // S - 4 + 1
#define TILE  64
#define TILES_PER_DIM 9   // ceil(515/64)
#define SA2   72          // LDS row stride in fp16 elems for a 64-wide K step (144B/row)

typedef __attribute__((ext_vector_type(16))) _Float16     v16h;
typedef __attribute__((ext_vector_type(8)))  float        v8f;
typedef __attribute__((ext_vector_type(4)))  unsigned int u32x4;
typedef __attribute__((ext_vector_type(2)))  unsigned int u32x2;

union Frag  { v16h v; u32x4 q[2]; };
union Pack4 { _Float16 h[4]; u32x2 u; };
union Pack8 { _Float16 h[8]; u32x4 u; };

__device__ __forceinline__ int clamp_row(int r) { return r < S_LEN ? r : (S_LEN - 1); }

// ---------- kernel 0: zero accumulators ----------
__global__ void zero_kernel(float* __restrict__ p, int n) {
  int i = blockIdx.x * blockDim.x + threadIdx.x;
  if (i < n) p[i] = 0.0f;
}

// ---------- kernel 1a (big-ws path): fused norms + fp16 hi/lo pre-conversion ----------
__global__ void __launch_bounds__(256) prep_kernel(const float* __restrict__ x1,
                                                   const float* __restrict__ x2,
                                                   _Float16* __restrict__ h1, _Float16* __restrict__ l1,
                                                   _Float16* __restrict__ h2, _Float16* __restrict__ l2,
                                                   float* __restrict__ n1, float* __restrict__ n2) {
  const int totalRows = B_LEN * S_LEN;
  int wave = (blockIdx.x * blockDim.x + threadIdx.x) >> 5;
  int lane = threadIdx.x & 31;
  const float* src; _Float16 *dh, *dl; float* dn; int row;
  if (wave < totalRows) { src = x1; dh = h1; dl = l1; dn = n1; row = wave; }
  else { row = wave - totalRows; if (row >= totalRows) return; src = x2; dh = h2; dl = l2; dn = n2; }
  const float4* r = (const float4*)(src + (size_t)row * D_LEN);
  _Float16* oh = dh + (size_t)row * D_LEN;
  _Float16* ol = dl + (size_t)row * D_LEN;
  float s = 0.0f;
#pragma unroll
  for (int w = 0; w < 4; ++w) {
    int i4 = w * 32 + lane;
    float4 v = r[i4];
    s += v.x * v.x + v.y * v.y + v.z * v.z + v.w * v.w;
    Pack4 ph, pl;
    ph.h[0] = (_Float16)v.x; ph.h[1] = (_Float16)v.y;
    ph.h[2] = (_Float16)v.z; ph.h[3] = (_Float16)v.w;
    pl.h[0] = (_Float16)(v.x - (float)ph.h[0]);
    pl.h[1] = (_Float16)(v.y - (float)ph.h[1]);
    pl.h[2] = (_Float16)(v.z - (float)ph.h[2]);
    pl.h[3] = (_Float16)(v.w - (float)ph.h[3]);
    *(u32x2*)(oh + (size_t)i4 * 4) = ph.u;
    *(u32x2*)(ol + (size_t)i4 * 4) = pl.u;
  }
#pragma unroll
  for (int m = 16; m; m >>= 1) s += __shfl_xor(s, m, 32);
  if (lane == 0) dn[row] = s;
}

// ---------- kernel 1b (small-ws path): norms only ----------
__global__ void __launch_bounds__(256) norms_kernel(const float* __restrict__ x1,
                                                    const float* __restrict__ x2,
                                                    float* __restrict__ n1,
                                                    float* __restrict__ n2) {
  const int totalRows = B_LEN * S_LEN;
  int wave = (blockIdx.x * blockDim.x + threadIdx.x) >> 5;
  int lane = threadIdx.x & 31;
  const float* src; float* dst; int row;
  if (wave < totalRows) { src = x1; dst = n1; row = wave; }
  else { row = wave - totalRows; if (row >= totalRows) return; src = x2; dst = n2; }
  const float4* r = (const float4*)(src + (size_t)row * D_LEN);
  float s = 0.0f;
#pragma unroll
  for (int w = 0; w < 4; ++w) {
    float4 v = r[w * 32 + lane];
    s += v.x * v.x + v.y * v.y + v.z * v.z + v.w * v.w;
  }
#pragma unroll
  for (int m = 16; m; m >>= 1) s += __shfl_xor(s, m, 32);
  if (lane == 0) dst[row] = s;
}

// ---------- shared GEMM core pieces ----------
struct GemmCtx {
  int i0, j0, tid, wave, waveM, waveN, r16, kh;
};

__device__ __forceinline__ GemmCtx make_ctx() {
  GemmCtx g;
  g.tid   = threadIdx.x;
  int lane = g.tid & 31;
  g.wave  = g.tid >> 5;
  g.waveM = g.wave >> 2;
  g.waveN = g.wave & 3;
  g.r16   = lane & 15;
  g.kh    = lane >> 4;
  g.i0    = (blockIdx.x / TILES_PER_DIM) * TILE;
  g.j0    = (blockIdx.x % TILES_PER_DIM) * TILE;
  return g;
}

__device__ __forceinline__ void wmma_chunk(const _Float16* aHi, const _Float16* aLo,
                                           const _Float16* bHi, const _Float16* bLo,
                                           int koff, const GemmCtx& g,
                                           v8f& acc0, v8f& acc1) {
  Frag ah0, al0, ah1, al1, bh, bl;
  int ar0 = (g.waveM * 32 + g.r16) * SA2 + koff;
  int ar1 = ar0 + 16 * SA2;
  int o0  = g.kh * 8;         // A-fragment: K 0..7 / 8..15
  int o1  = 16 + g.kh * 8;    //             K 16..23 / 24..31
  ah0.q[0] = *(const u32x4*)&aHi[ar0 + o0]; ah0.q[1] = *(const u32x4*)&aHi[ar0 + o1];
  al0.q[0] = *(const u32x4*)&aLo[ar0 + o0]; al0.q[1] = *(const u32x4*)&aLo[ar0 + o1];
  ah1.q[0] = *(const u32x4*)&aHi[ar1 + o0]; ah1.q[1] = *(const u32x4*)&aHi[ar1 + o1];
  al1.q[0] = *(const u32x4*)&aLo[ar1 + o0]; al1.q[1] = *(const u32x4*)&aLo[ar1 + o1];
  int bc = (g.waveN * 16 + g.r16) * SA2 + koff;  // B-fragment: same K order as A
  bh.q[0] = *(const u32x4*)&bHi[bc + o0]; bh.q[1] = *(const u32x4*)&bHi[bc + o1];
  bl.q[0] = *(const u32x4*)&bLo[bc + o0]; bl.q[1] = *(const u32x4*)&bLo[bc + o1];
  // fp16x3: hi*hi + hi*lo + lo*hi  (lo*lo negligible)
  acc0 = __builtin_amdgcn_wmma_f32_16x16x32_f16(false, ah0.v, false, bh.v, (short)0, acc0, false, false);
  acc0 = __builtin_amdgcn_wmma_f32_16x16x32_f16(false, ah0.v, false, bl.v, (short)0, acc0, false, false);
  acc0 = __builtin_amdgcn_wmma_f32_16x16x32_f16(false, al0.v, false, bh.v, (short)0, acc0, false, false);
  acc1 = __builtin_amdgcn_wmma_f32_16x16x32_f16(false, ah1.v, false, bh.v, (short)0, acc1, false, false);
  acc1 = __builtin_amdgcn_wmma_f32_16x16x32_f16(false, ah1.v, false, bl.v, (short)0, acc1, false, false);
  acc1 = __builtin_amdgcn_wmma_f32_16x16x32_f16(false, al1.v, false, bh.v, (short)0, acc1, false, false);
}

__device__ __forceinline__ void epilogue(float* cTile, float* nRow, float* nCol,
                                         const float* n1b, const float* n2b,
                                         float* v1b, float* v2b,
                                         const GemmCtx& g, v8f& acc0, v8f& acc1) {
  __syncthreads();
  if (g.tid < TILE) {
    int i = g.i0 + g.tid;
    nRow[g.tid] = (i < S_LEN) ? n1b[i] : 0.0f;
  } else if (g.tid < 2 * TILE) {
    int j = g.j0 + (g.tid - TILE);
    nCol[g.tid - TILE] = (j < S_LEN) ? n2b[j] : 0.0f;
  }
  __syncthreads();
  {
    const int col = g.waveN * 16 + g.r16;
    const int gj  = g.j0 + col;
    const float nb = nCol[col];
#pragma unroll
    for (int r = 0; r < 8; ++r) {
      int rl0 = g.waveM * 32 + g.kh * 8 + r;   // acc0 row per C/D layout
      int rl1 = rl0 + 16;                      // acc1 row
      float a0 = 0.0f, a1 = 0.0f;
      if ((g.i0 + rl0) < S_LEN && gj < S_LEN) {
        float sq = nRow[rl0] + nb - 2.0f * acc0[r];
        sq = sq > 0.0f ? sq : 0.0f;
        a0 = 1.0f / (1.0f + sqrtf(sq));
      }
      if ((g.i0 + rl1) < S_LEN && gj < S_LEN) {
        float sq = nRow[rl1] + nb - 2.0f * acc1[r];
        sq = sq > 0.0f ? sq : 0.0f;
        a1 = 1.0f / (1.0f + sqrtf(sq));
      }
      cTile[rl0 * 65 + col] = a0;
      cTile[rl1 * 65 + col] = a1;
    }
  }
  __syncthreads();
  if (g.tid < TILE) {                       // row sums -> x1_vec
    int i = g.i0 + g.tid;
    if (i < S_LEN) {
      float s = 0.0f;
#pragma unroll 8
      for (int c = 0; c < TILE; ++c) s += cTile[g.tid * 65 + c];
      atomicAdd(&v1b[i], s);
    }
  } else if (g.tid < 2 * TILE) {            // col sums -> x2_vec
    int j = g.j0 + (g.tid - TILE);
    if (j < S_LEN) {
      float s = 0.0f;
#pragma unroll 8
      for (int c = 0; c < TILE; ++c) s += cTile[c * 65 + (g.tid - TILE)];
      atomicAdd(&v2b[j], s);
    }
  }
}

// ---------- kernel 2a (big-ws path): pipelined GEMM on pre-converted fp16 hi/lo ----------
__global__ void __launch_bounds__(256) dist_gemm_pre_kernel(const _Float16* __restrict__ h1,
                                                            const _Float16* __restrict__ l1,
                                                            const _Float16* __restrict__ h2,
                                                            const _Float16* __restrict__ l2,
                                                            const float* __restrict__ n1,
                                                            const float* __restrict__ n2,
                                                            float* __restrict__ v1,
                                                            float* __restrict__ v2) {
  __shared__ __align__(16) _Float16 aHi[TILE * SA2], aLo[TILE * SA2];
  __shared__ __align__(16) _Float16 bHi[TILE * SA2], bLo[TILE * SA2];
  __shared__ float cTile[TILE * 65];
  __shared__ float nRow[TILE], nCol[TILE];

  GemmCtx g = make_ctx();
  const int b = blockIdx.y;
  const size_t bofs = (size_t)b * S_LEN * D_LEN;
  const _Float16* h1b = h1 + bofs; const _Float16* l1b = l1 + bofs;
  const _Float16* h2b = h2 + bofs; const _Float16* l2b = l2 + bofs;
  const float* n1b = n1 + (size_t)b * S_LEN;
  const float* n2b = n2 + (size_t)b * S_LEN;
  float* v1b = v1 + (size_t)b * S_LEN;
  float* v2b = v2 + (size_t)b * S_LEN;

  // Per-thread staging: rows r0 (0..31) and r1 (32..63), one 8-halfword group each.
  // Rows are CLAMPED (not predicated): padded rows feed WMMA results that the
  // epilogue masks out, so branchless loads are safe.
  const int r0 = g.tid >> 3, r1 = r0 + 32;
  const int c8 = (g.tid & 7) * 8;
  const size_t gA0 = (size_t)clamp_row(g.i0 + r0) * D_LEN + c8;
  const size_t gA1 = (size_t)clamp_row(g.i0 + r1) * D_LEN + c8;
  const size_t gB0 = (size_t)clamp_row(g.j0 + r0) * D_LEN + c8;
  const size_t gB1 = (size_t)clamp_row(g.j0 + r1) * D_LEN + c8;
  const int l0 = r0 * SA2 + c8, l1o = r1 * SA2 + c8;

  u32x4 sAh0, sAh1, sAl0, sAl1, sBh0, sBh1, sBl0, sBl1;
  auto loadRegs = [&](int k) {
    sAh0 = *(const u32x4*)(h1b + gA0 + k); sAh1 = *(const u32x4*)(h1b + gA1 + k);
    sAl0 = *(const u32x4*)(l1b + gA0 + k); sAl1 = *(const u32x4*)(l1b + gA1 + k);
    sBh0 = *(const u32x4*)(h2b + gB0 + k); sBh1 = *(const u32x4*)(h2b + gB1 + k);
    sBl0 = *(const u32x4*)(l2b + gB0 + k); sBl1 = *(const u32x4*)(l2b + gB1 + k);
  };

  v8f acc0 = {}; v8f acc1 = {};
  loadRegs(0);
  for (int k0 = 0; k0 < D_LEN; k0 += 64) {
    __syncthreads();                     // previous iteration's fragment reads done
    *(u32x4*)&aHi[l0]  = sAh0; *(u32x4*)&aHi[l1o] = sAh1;
    *(u32x4*)&aLo[l0]  = sAl0; *(u32x4*)&aLo[l1o] = sAl1;
    *(u32x4*)&bHi[l0]  = sBh0; *(u32x4*)&bHi[l1o] = sBh1;
    *(u32x4*)&bLo[l0]  = sBl0; *(u32x4*)&bLo[l1o] = sBl1;
    __syncthreads();
    if (k0 + 64 < D_LEN) loadRegs(k0 + 64);   // overlap next loads with WMMA
    wmma_chunk(aHi, aLo, bHi, bLo, 0,  g, acc0, acc1);
    wmma_chunk(aHi, aLo, bHi, bLo, 32, g, acc0, acc1);
  }
  epilogue(cTile, nRow, nCol, n1b, n2b, v1b, v2b, g, acc0, acc1);
}

// ---------- kernel 2b (small-ws path): pipelined GEMM with in-kernel fp16 split ----------
__global__ void __launch_bounds__(256) dist_gemm_cvt_kernel(const float* __restrict__ x1,
                                                            const float* __restrict__ x2,
                                                            const float* __restrict__ n1,
                                                            const float* __restrict__ n2,
                                                            float* __restrict__ v1,
                                                            float* __restrict__ v2) {
  __shared__ __align__(16) _Float16 aHi[TILE * SA2], aLo[TILE * SA2];
  __shared__ __align__(16) _Float16 bHi[TILE * SA2], bLo[TILE * SA2];
  __shared__ float cTile[TILE * 65];
  __shared__ float nRow[TILE], nCol[TILE];

  GemmCtx g = make_ctx();
  const int b = blockIdx.y;
  const float* Ab  = x1 + (size_t)b * S_LEN * D_LEN;
  const float* Bb  = x2 + (size_t)b * S_LEN * D_LEN;
  const float* n1b = n1 + (size_t)b * S_LEN;
  const float* n2b = n2 + (size_t)b * S_LEN;
  float* v1b = v1 + (size_t)b * S_LEN;
  float* v2b = v2 + (size_t)b * S_LEN;

  const int r0 = g.tid >> 3, r1 = r0 + 32;
  const int c8 = (g.tid & 7) * 8;
  const size_t gA0 = (size_t)clamp_row(g.i0 + r0) * D_LEN + c8;
  const size_t gA1 = (size_t)clamp_row(g.i0 + r1) * D_LEN + c8;
  const size_t gB0 = (size_t)clamp_row(g.j0 + r0) * D_LEN + c8;
  const size_t gB1 = (size_t)clamp_row(g.j0 + r1) * D_LEN + c8;
  const int l0 = r0 * SA2 + c8, l1o = r1 * SA2 + c8;

  Pack8 hA0, lA0, hA1, lA1, hB0, lB0, hB1, lB1;
  auto cvt8 = [](const float* p, Pack8& hi, Pack8& lo) {
    float4 u = *(const float4*)p;
    float4 v = *(const float4*)(p + 4);
    hi.h[0] = (_Float16)u.x; hi.h[1] = (_Float16)u.y;
    hi.h[2] = (_Float16)u.z; hi.h[3] = (_Float16)u.w;
    hi.h[4] = (_Float16)v.x; hi.h[5] = (_Float16)v.y;
    hi.h[6] = (_Float16)v.z; hi.h[7] = (_Float16)v.w;
    lo.h[0] = (_Float16)(u.x - (float)hi.h[0]);
    lo.h[1] = (_Float16)(u.y - (float)hi.h[1]);
    lo.h[2] = (_Float16)(u.z - (float)hi.h[2]);
    lo.h[3] = (_Float16)(u.w - (float)hi.h[3]);
    lo.h[4] = (_Float16)(v.x - (float)hi.h[4]);
    lo.h[5] = (_Float16)(v.y - (float)hi.h[5]);
    lo.h[6] = (_Float16)(v.z - (float)hi.h[6]);
    lo.h[7] = (_Float16)(v.w - (float)hi.h[7]);
  };
  auto loadRegs = [&](int k) {
    cvt8(Ab + gA0 + k, hA0, lA0);
    cvt8(Ab + gA1 + k, hA1, lA1);
    cvt8(Bb + gB0 + k, hB0, lB0);
    cvt8(Bb + gB1 + k, hB1, lB1);
  };

  v8f acc0 = {}; v8f acc1 = {};
  loadRegs(0);
  for (int k0 = 0; k0 < D_LEN; k0 += 64) {
    __syncthreads();
    *(u32x4*)&aHi[l0]  = hA0.u; *(u32x4*)&aHi[l1o] = hA1.u;
    *(u32x4*)&aLo[l0]  = lA0.u; *(u32x4*)&aLo[l1o] = lA1.u;
    *(u32x4*)&bHi[l0]  = hB0.u; *(u32x4*)&bHi[l1o] = hB1.u;
    *(u32x4*)&bLo[l0]  = lB0.u; *(u32x4*)&bLo[l1o] = lB1.u;
    __syncthreads();
    if (k0 + 64 < D_LEN) loadRegs(k0 + 64);   // conversion overlaps WMMA
    wmma_chunk(aHi, aLo, bHi, bLo, 0,  g, acc0, acc1);
    wmma_chunk(aHi, aLo, bHi, bLo, 32, g, acc0, acc1);
  }
  epilogue(cTile, nRow, nCol, n1b, n2b, v1b, v2b, g, acc0, acc1);
}

// ---------- kernel 3: attention-weighted window pooling (width 4) ----------
__global__ void __launch_bounds__(256) pool_kernel(const float* __restrict__ x1,
                                                   const float* __restrict__ x2,
                                                   const float* __restrict__ v1,
                                                   const float* __restrict__ v2,
                                                   float* __restrict__ out) {
  __shared__ __align__(16) float rows[19 * D_LEN];
  __shared__ float wv[19];
  const int b     = blockIdx.y;
  const int j0    = blockIdx.x * 16;
  const int which = blockIdx.z;
  const float* x = which ? x2 : x1;
  const float* v = which ? v2 : v1;
  float* o = out + (size_t)which * B_LEN * SOUT * D_LEN;

  const float* xb = x + (size_t)b * S_LEN * D_LEN;
  const float* vb = v + (size_t)b * S_LEN;
  const int tid = threadIdx.x;

  if (tid < 19) wv[tid] = vb[j0 + tid];
  for (int p = tid; p < 19 * 128; p += 256) {
    int rr = p >> 7, c4 = p & 127;
    *(float4*)&rows[rr * D_LEN + c4 * 4] =
        *(const float4*)(xb + (size_t)(j0 + rr) * D_LEN + c4 * 4);
  }
  __syncthreads();

  float* ob = o + ((size_t)b * SOUT + j0) * D_LEN;
  for (int p = tid; p < 16 * 128; p += 256) {
    int j = p >> 7, c4 = p & 127;
    float4 acc = make_float4(0.f, 0.f, 0.f, 0.f);
#pragma unroll
    for (int k = 0; k < 4; ++k) {
      float w = wv[j + k];
      float4 r = *(const float4*)&rows[(j + k) * D_LEN + c4 * 4];
      acc.x += w * r.x; acc.y += w * r.y; acc.z += w * r.z; acc.w += w * r.w;
    }
    *(float4*)(ob + (size_t)j * D_LEN + c4 * 4) = acc;
  }
}

// ---------- launch ----------
extern "C" void kernel_launch(void* const* d_in, const int* in_sizes, int n_in,
                              void* d_out, int out_size, void* d_ws, size_t ws_size,
                              hipStream_t stream) {
  const float* x1 = (const float*)d_in[0];
  const float* x2 = (const float*)d_in[1];
  float* ws = (float*)d_ws;
  const int BS = B_LEN * S_LEN;
  const size_t elems = (size_t)B_LEN * S_LEN * D_LEN;

  float* n1 = ws;
  float* n2 = n1 + BS;
  float* v1 = n2 + BS;
  float* v2 = v1 + BS;

  // zero the atomic accumulators (v1, v2 contiguous)
  int zn = 2 * BS;
  zero_kernel<<<(zn + 255) / 256, 256, 0, stream>>>(v1, zn);

  const size_t head = (size_t)4 * BS * sizeof(float);
  const size_t need = head + 4 * elems * sizeof(_Float16);
  const int wave_blocks = (2 * BS) / 8;   // one wave32 per row, 8 waves per block
  dim3 g(TILES_PER_DIM * TILES_PER_DIM, B_LEN);

  if (ws_size >= need) {
    // Big-workspace path: convert once, stream fp16 through the GEMM.
    _Float16* h1 = (_Float16*)((char*)d_ws + head);
    _Float16* l1 = h1 + elems;
    _Float16* h2 = l1 + elems;
    _Float16* l2 = h2 + elems;
    prep_kernel<<<wave_blocks, 256, 0, stream>>>(x1, x2, h1, l1, h2, l2, n1, n2);
    dist_gemm_pre_kernel<<<g, 256, 0, stream>>>(h1, l1, h2, l2, n1, n2, v1, v2);
  } else {
    // Small-workspace path: native fp16 split inside the GEMM.
    norms_kernel<<<wave_blocks, 256, 0, stream>>>(x1, x2, n1, n2);
    dist_gemm_cvt_kernel<<<g, 256, 0, stream>>>(x1, x2, n1, n2, v1, v2);
  }

  // pooling for both outputs (z = 0 -> w1, z = 1 -> w2)
  dim3 gp(SOUT / 16, B_LEN, 2);
  pool_kernel<<<gp, 256, 0, stream>>>(x1, x2, v1, v2, (float*)d_out);
}
